// DiscriminativeLoss_163208757493
// MI455X (gfx1250) — matrix operand order repaired and checked by reading, not compiled, not run on hardware
//
#include <hip/hip_runtime.h>
#include <hip/hip_bf16.h>

typedef __attribute__((ext_vector_type(16))) _Float16 v16h;
typedef __attribute__((ext_vector_type(2)))  _Float16 h2;
typedef __attribute__((ext_vector_type(2)))  __fp16   fp16x2;
typedef __attribute__((ext_vector_type(8)))  float    v8f;

#if defined(__AMDGCN__) && __has_builtin(__builtin_amdgcn_tensor_load_to_lds)
#define DL_TDM 1
#else
#define DL_TDM 0
#endif

#if defined(__AMDGCN__) && __has_builtin(__builtin_amdgcn_ballot_w32)
#define DL_BALLOT32(p) __builtin_amdgcn_ballot_w32(p)
#else
#define DL_BALLOT32(p) ((unsigned)__ballot(p))
#endif

namespace dl {
constexpr int   B       = 32;
constexpr int   C       = 8;
constexpr int   P       = 256 * 512;     // pixels per sample
constexpr int   K       = 5;
constexpr float DELTA_V = 0.5f;
constexpr float DELTA_D = 3.0f;
constexpr float P_VAR   = 1.0f;
constexpr float P_DIST  = 1.0f;
constexpr float P_REG   = 0.001f;

// phase-1 tiling (WMMA segmented sum)
constexpr int BPS  = 32;                  // blocks per sample
constexpr int WPB  = 8;                   // waves per block (256 thr, wave32)
constexpr int PPW  = P / (BPS * WPB);     // 512 pixels per wave
constexpr int IT1  = PPW / 32;            // 16 WMMA tiles per wave

// phase-3 tiling (pull term)
constexpr int BPS3 = 32;                  // blocks per sample
constexpr int PPB3 = P / BPS3;            // 4096 pixels per block
constexpr int TILE3 = 256;                // pixels per TDM tile (1 per thread)
constexpr int IT3  = PPB3 / TILE3;        // 16 tiles per block

// workspace layout (float offsets)
constexpr int WS_SUM = 0;                 // [B][40]  per-cluster channel sums
constexpr int WS_CNT = WS_SUM + B * 40;   // [B][8]   per-cluster counts
constexpr int WS_MU  = WS_CNT + B * 8;    // [B][40]  cluster means
constexpr int WS_VS  = WS_MU  + B * 40;   // [B][8]   pull-term partial sums
constexpr int WS_LD  = WS_VS  + B * 8;    // [B]      per-sample l_dist
constexpr int WS_LR  = WS_LD  + B;        // [B]      per-sample l_reg
constexpr int WS_TOT = WS_LR  + B;
} // namespace dl

union V16u { v16h h; h2 p[8]; fp16x2 q[8]; unsigned int u[8]; };

__global__ void dl_zero(float* __restrict__ ws, int n) {
  for (int i = threadIdx.x; i < n; i += blockDim.x) ws[i] = 0.0f;
}

// ---------------------------------------------------------------------------
// Phase 1: segmented sum via WMMA.  D[cluster][channel] += onehot x feat.
//   A (16x32 f16): one-hot built from wave ballots, M = cluster, K = pixel.
//   B (32x16 f16): feature tile, K = pixel slot, N = channel (cols 0..7 used).
//   Lanes n>=8 feed junk (clamped channel-0 data) into D columns 8..15, which
//   are never read: WMMA columns are independent, so no masking is needed.
// ---------------------------------------------------------------------------
__global__ void dl_phase1(const float* __restrict__ feat,
                          const int*   __restrict__ label,
                          float*       __restrict__ ws) {
  using namespace dl;
  const int b    = blockIdx.x / BPS;
  const int blk  = blockIdx.x % BPS;
  const int wave = threadIdx.x >> 5;
  const int lane = threadIdx.x & 31;

  const float* featb = feat  + (size_t)b * C * P;
  const int*   labb  = label + (size_t)b * P;
  const int    pbase = (blk * WPB + wave) * PPW;

  // A-matrix lane mapping (16-bit A 16x32, contiguous k = 16*(lane/16) + s, as in B)
  const int m      = lane & 15;               // cluster row
  const int kbaseA = (lane < 16) ? 0 : 16;
  // Loop-invariant lane predicates -> hoisted v_cmp, in-loop v_cndmask only.
  const bool is0 = (m == 0), is1 = (m == 1), is2 = (m == 2);
  const bool is3 = (m == 3), is4 = (m == 4);
  // B-matrix lane mapping (16-bit B 32x16, row striped across lanes)
  const int  n      = lane & 15;              // channel column
  const int  kbaseB = (lane < 16) ? 0 : 16;
  const float* fchan = featb + (size_t)((n < C) ? n : 0) * P;  // clamp: in-bounds

  v8f acc = {};
  int cn0 = 0, cn1 = 0, cn2 = 0, cn3 = 0, cn4 = 0;   // wave-uniform counts

  for (int it = 0; it < IT1; ++it) {
    const int p0  = pbase + it * 32;
    const int lab = labb[p0 + lane];          // one label per lane (coalesced)

    // Wave32 ballots: bit i of mk <=> pixel (p0+i) belongs to cluster k.
    const unsigned m0 = DL_BALLOT32(lab == 0);
    const unsigned m1 = DL_BALLOT32(lab == 1);
    const unsigned m2 = DL_BALLOT32(lab == 2);
    const unsigned m3 = DL_BALLOT32(lab == 3);
    const unsigned m4 = DL_BALLOT32(lab == 4);
    cn0 += __popc(m0); cn1 += __popc(m1); cn2 += __popc(m2);
    cn3 += __popc(m3); cn4 += __popc(m4);

    // Branch-free mask selection: at most one predicate hits per lane.
    const unsigned vm = (is0 ? m0 : 0u) | (is1 ? m1 : 0u) | (is2 ? m2 : 0u) |
                        (is3 ? m3 : 0u) | (is4 ? m4 : 0u);
    const unsigned lo = (vm >> kbaseA) & 0xFFu;         // K slots base..base+7
    const unsigned hi = (vm >> (kbaseA + 8)) & 0xFFu;   // K slots base+8..+15

    V16u A, Bm;
#pragma unroll
    for (int v = 0; v < 4; ++v) {
      const unsigned p2 = (lo >> (2 * v)) & 3u;
      A.u[v] = ((p2 & 1u) ? 0x3C00u : 0u) | ((p2 & 2u) ? 0x3C000000u : 0u);
    }
#pragma unroll
    for (int v = 0; v < 4; ++v) {
      const unsigned p2 = (hi >> (2 * v)) & 3u;
      A.u[4 + v] = ((p2 & 1u) ? 0x3C00u : 0u) | ((p2 & 2u) ? 0x3C000000u : 0u);
    }

    // Feature tile: 16 consecutive f32 per lane -> 4x b128 loads, pack to f16.
    const float4* fq = reinterpret_cast<const float4*>(fchan + p0 + kbaseB);
    const float4 q0 = fq[0], q1 = fq[1], q2 = fq[2], q3 = fq[3];
    Bm.q[0] = __builtin_amdgcn_cvt_pkrtz(q0.x, q0.y);
    Bm.q[1] = __builtin_amdgcn_cvt_pkrtz(q0.z, q0.w);
    Bm.q[2] = __builtin_amdgcn_cvt_pkrtz(q1.x, q1.y);
    Bm.q[3] = __builtin_amdgcn_cvt_pkrtz(q1.z, q1.w);
    Bm.q[4] = __builtin_amdgcn_cvt_pkrtz(q2.x, q2.y);
    Bm.q[5] = __builtin_amdgcn_cvt_pkrtz(q2.z, q2.w);
    Bm.q[6] = __builtin_amdgcn_cvt_pkrtz(q3.x, q3.y);
    Bm.q[7] = __builtin_amdgcn_cvt_pkrtz(q3.z, q3.w);

    if (it + 1 < IT1)
      __builtin_prefetch(fchan + p0 + 32 + kbaseB, 0, 1);  // global_prefetch_b8

    acc = __builtin_amdgcn_wmma_f32_16x16x32_f16(
        /*neg_a=*/false, A.h, /*neg_b=*/false, Bm.h,
        /*c_mod=*/(short)0, acc, /*reuse_a=*/false, /*reuse_b=*/false);
  }

  __shared__ float ssum[WPB][40];
  __shared__ float scnt[WPB][K];
  union { v8f v; float f[8]; } au;
  au.v = acc;
  // C/D layout: VGPR r, lanes 0..15 -> M=r, N=lane. Useful: r<5, lane<8.
  if (lane < 8) {
#pragma unroll
    for (int r = 0; r < K; ++r) ssum[wave][r * 8 + lane] = au.f[r];
  }
  if (lane == 0) {   // counts are wave-uniform (SALU popcounts)
    scnt[wave][0] = (float)cn0; scnt[wave][1] = (float)cn1;
    scnt[wave][2] = (float)cn2; scnt[wave][3] = (float)cn3;
    scnt[wave][4] = (float)cn4;
  }
  __syncthreads();

  if (threadIdx.x < 40) {
    float t = 0.f;
#pragma unroll
    for (int w = 0; w < WPB; ++w) t += ssum[w][threadIdx.x];
    atomicAdd(&ws[WS_SUM + b * 40 + threadIdx.x], t);
  }
  if (threadIdx.x < K) {
    float t = 0.f;
#pragma unroll
    for (int w = 0; w < WPB; ++w) t += scnt[w][threadIdx.x];
    atomicAdd(&ws[WS_CNT + b * 8 + threadIdx.x], t);
  }
}

// ---------------------------------------------------------------------------
// Phase 2: mu = sum/count; l_dist (pairwise push) and l_reg per sample.
// ---------------------------------------------------------------------------
__global__ void dl_phase2(float* __restrict__ ws) {
  using namespace dl;
  const int b = threadIdx.x;
  if (b >= B) return;

  float cnt[K];
  float mu[K][C];
#pragma unroll
  for (int k = 0; k < K; ++k) cnt[k] = fmaxf(ws[WS_CNT + b * 8 + k], 1.0f);
#pragma unroll
  for (int k = 0; k < K; ++k)
#pragma unroll
    for (int c = 0; c < C; ++c) {
      const float m = ws[WS_SUM + b * 40 + k * 8 + c] / cnt[k];
      mu[k][c] = m;
      ws[WS_MU + b * 40 + k * 8 + c] = m;
    }

  float lr = 0.f;
#pragma unroll
  for (int k = 0; k < K; ++k) {
    float s = 0.f;
#pragma unroll
    for (int c = 0; c < C; ++c) s += mu[k][c] * mu[k][c];
    lr += sqrtf(s);
  }
  lr *= (1.0f / (float)K);

  float ld = 0.f;
#pragma unroll
  for (int i = 0; i < K; ++i)
#pragma unroll
    for (int j = 0; j < K; ++j) {
      if (i == j) continue;
      float sq = 0.f;
#pragma unroll
      for (int c = 0; c < C; ++c) {
        const float d = mu[i][c] - mu[j][c];
        sq += d * d;
      }
      const float t = fmaxf(2.0f * DELTA_D - sqrtf(sq), 0.0f);
      ld += t * t;
    }
  ld *= (1.0f / (float)(K * K - K));

  ws[WS_LD + b] = ld;
  ws[WS_LR + b] = lr;
}

// ---------------------------------------------------------------------------
// TDM helper: 2D tile (tileX pixels x tileY channels, f32) global -> LDS.
// D# built per ISA 8.3/8.4: group0 = {count/type, lds_addr, global_addr},
// group1 = {data_size, tensor dims, tile dims, dim0 stride}.
// ---------------------------------------------------------------------------
#if DL_TDM
typedef __attribute__((ext_vector_type(4))) unsigned int u32x4;
typedef __attribute__((ext_vector_type(8))) int          i32x8;
typedef __attribute__((ext_vector_type(4))) int          i32x4;

__device__ __forceinline__ void tdm_load_tile(unsigned lds_off,
                                              const float* gptr,
                                              int tileX, int tileY,
                                              long long strideElems) {
  const unsigned long long ga = (unsigned long long)(size_t)gptr;
  const unsigned long long st = (unsigned long long)strideElems;
  u32x4 g0;
  g0[0] = 1u;                                        // count=1, user descriptor
  g0[1] = lds_off;                                   // lds_addr (bytes)
  g0[2] = (unsigned)ga;                              // global_addr[31:0]
  g0[3] = ((unsigned)(ga >> 32) & 0x01FFFFFFu) | 0x80000000u; // addr[56:32]|type=2
  i32x8 g1;
  g1[0] = (int)0x00020000u;                          // data_size = 4 bytes
  g1[1] = (int)(((unsigned)tileX & 0xFFFFu) << 16);  // tensor_dim0[15:0]
  g1[2] = (int)((((unsigned)tileX >> 16) & 0xFFFFu) |
                (((unsigned)tileY & 0xFFFFu) << 16));  // td0 hi | td1 lo
  g1[3] = (int)((((unsigned)tileY >> 16) & 0xFFFFu) |
                (((unsigned)tileX & 0xFFFFu) << 16));  // td1 hi | tile_dim0
  g1[4] = (int)((unsigned)tileY & 0xFFFFu);          // tile_dim1 (tile_dim2=0)
  g1[5] = (int)(unsigned)(st & 0xFFFFFFFFull);       // dim0 stride[31:0]
  g1[6] = (int)(unsigned)((st >> 32) & 0xFFFFull);   // dim0 stride[47:32]
  g1[7] = 0;
  i32x4 g2 = {0, 0, 0, 0};
  i32x4 g3 = {0, 0, 0, 0};
#if __has_include(<hip/amd_detail/amd_gfx1250_TDM.h>)
  i32x8 g4 = {0, 0, 0, 0, 0, 0, 0, 0};
  __builtin_amdgcn_tensor_load_to_lds(g0, g1, g2, g3, g4, 0);
#else
  __builtin_amdgcn_tensor_load_to_lds(g0, g1, g2, g3, 0);
#endif
}
#endif // DL_TDM

// ---------------------------------------------------------------------------
// Phase 3: pull term. Feature tiles staged into LDS by the Tensor Data Mover
// (double-buffered, s_wait_tensorcnt), compute reads LDS; falls back to
// direct global loads if the TDM builtin is unavailable.
// ---------------------------------------------------------------------------
__global__ void dl_phase3(const float* __restrict__ feat,
                          const int*   __restrict__ label,
                          float*       __restrict__ ws) {
  using namespace dl;
  const int b    = blockIdx.x / BPS3;
  const int blk  = blockIdx.x % BPS3;
  const int tid  = threadIdx.x;
  const int wave = tid >> 5;
  (void)wave;

  __shared__ float smu[K * C];
  __shared__ float red[256];
  if (tid < K * C) smu[tid] = ws[WS_MU + b * 40 + tid];

  const float* featb = feat  + (size_t)b * C * P;
  const int*   labb  = label + (size_t)b * P;
  const int    base  = blk * PPB3;

  float s[K] = {0.f, 0.f, 0.f, 0.f, 0.f};

#if DL_TDM
  __shared__ float sbuf[2][C * TILE3];     // 2 x 8 KB double buffer
  const unsigned lds0 = (unsigned)(size_t)&sbuf[0][0];
  const unsigned lds1 = (unsigned)(size_t)&sbuf[1][0];
  if (wave == 0)                           // TDM ignores EXEC; scalar-gated
    tdm_load_tile(lds0, featb + base, TILE3, C, (long long)P);
  __syncthreads();

  for (int i = 0; i < IT3; ++i) {
    if (wave == 0) {
      if (i + 1 < IT3) {
        tdm_load_tile(((i + 1) & 1) ? lds1 : lds0,
                      featb + base + (i + 1) * TILE3, TILE3, C, (long long)P);
        __builtin_amdgcn_s_wait_tensorcnt(1);   // tile i complete (in-order)
      } else {
        __builtin_amdgcn_s_wait_tensorcnt(0);
      }
    }
    __syncthreads();                       // tile i visible to all waves

    const float* sb  = sbuf[i & 1];
    const int    p   = base + i * TILE3 + tid;
    const int    lab = labb[p];
    float d2 = 0.f;
#pragma unroll
    for (int c = 0; c < C; ++c) {
      const float diff = smu[lab * C + c] - sb[c * TILE3 + tid];
      d2 += diff * diff;
    }
    const float t  = fmaxf(sqrtf(d2) - DELTA_V, 0.0f);
    const float t2 = t * t;
#pragma unroll
    for (int k = 0; k < K; ++k) s[k] += (lab == k) ? t2 : 0.0f;
    __syncthreads();                       // buffer reusable
  }
#else
  __syncthreads();
  for (int i = 0; i < IT3; ++i) {
    const int p   = base + i * TILE3 + tid;
    const int lab = labb[p];
    float d2 = 0.f;
#pragma unroll
    for (int c = 0; c < C; ++c) {
      const float diff = smu[lab * C + c] - featb[(size_t)c * P + p];
      d2 += diff * diff;
    }
    const float t  = fmaxf(sqrtf(d2) - DELTA_V, 0.0f);
    const float t2 = t * t;
#pragma unroll
    for (int k = 0; k < K; ++k) s[k] += (lab == k) ? t2 : 0.0f;
    if (i + 1 < IT3)
      __builtin_prefetch(featb + base + (i + 1) * TILE3 + tid, 0, 1);
  }
#endif

#pragma unroll
  for (int k = 0; k < K; ++k) {
    red[tid] = s[k];
    __syncthreads();
    for (int off = 128; off > 0; off >>= 1) {
      if (tid < off) red[tid] += red[tid + off];
      __syncthreads();
    }
    if (tid == 0) atomicAdd(&ws[WS_VS + b * 8 + k], red[0]);
    __syncthreads();
  }
}

// ---------------------------------------------------------------------------
// Phase 4: finalize the four scalars.
// ---------------------------------------------------------------------------
__global__ void dl_phase4(const float* __restrict__ ws, float* __restrict__ out) {
  using namespace dl;
  const int b = threadIdx.x;
  __shared__ float slv[B], sld[B], slr[B];
  if (b < B) {
    float lv = 0.f;
#pragma unroll
    for (int k = 0; k < K; ++k)
      lv += ws[WS_VS + b * 8 + k] / fmaxf(ws[WS_CNT + b * 8 + k], 1.0f);
    slv[b] = lv * (1.0f / (float)K);
    sld[b] = ws[WS_LD + b];
    slr[b] = ws[WS_LR + b];
  }
  __syncthreads();
  if (b == 0) {
    float a = 0.f, d = 0.f, r = 0.f;
    for (int i = 0; i < B; ++i) { a += slv[i]; d += sld[i]; r += slr[i]; }
    const float lvr = P_VAR  * a / (float)B;
    const float ldr = P_DIST * d / (float)B;
    const float lrr = P_REG  * r / (float)B;
    out[0] = lvr + ldr + lrr;
    out[1] = lvr;
    out[2] = ldr;
    out[3] = lrr;
  }
}

extern "C" void kernel_launch(void* const* d_in, const int* in_sizes, int n_in,
                              void* d_out, int out_size, void* d_ws, size_t ws_size,
                              hipStream_t stream) {
  using namespace dl;
  (void)in_sizes; (void)n_in; (void)out_size; (void)ws_size;
  const float* feat  = (const float*)d_in[0];
  const int*   label = (const int*)d_in[1];   // labels in [0,5): low 32 bits suffice
  float*       out   = (float*)d_out;
  float*       ws    = (float*)d_ws;

  dl_zero  <<<1,          256, 0, stream>>>(ws, WS_TOT);
  dl_phase1<<<B * BPS,    256, 0, stream>>>(feat, label, ws);
  dl_phase2<<<1,           32, 0, stream>>>(ws);
  dl_phase3<<<B * BPS3,   256, 0, stream>>>(feat, label, ws);
  dl_phase4<<<1,           32, 0, stream>>>(ws, out);
}
